// CRF_60378650247194
// MI455X (gfx1250) — matrix operand order repaired and checked by reading, not compiled, not run on hardware
//
#include <hip/hip_runtime.h>
#include <hip/hip_bf16.h>
#include <math.h>

#define NTS 128          // number of tags/states
#define TLEN 262144      // sequence length
#define CHUNK 256        // time steps per chunk (matrices per workgroup)
#define NCHUNK 1024      // chunks covering t = 1 .. TLEN-1 (last chunk has 255)
#define SPB 136          // padded LDS row stride in bf16 elements (272 B)
#define SPF 130          // padded f32 LDS stride for the combine kernel
#define NSCORE 512       // blocks for score reduction

typedef __attribute__((ext_vector_type(16))) __bf16 v16bf;
typedef __attribute__((ext_vector_type(8)))  __bf16 v8bf;
typedef __attribute__((ext_vector_type(8)))  float  v8f;
typedef __attribute__((ext_vector_type(2)))  unsigned short us2;

__device__ __forceinline__ float bf2f(unsigned short h) {
  unsigned u = ((unsigned)h) << 16;
  return __builtin_bit_cast(float, u);
}
__device__ __forceinline__ unsigned short bfbits(float f) {
  return __builtin_bit_cast(unsigned short, (__bf16)f);
}

// row-max of 128 non-negative bf16 values via packed u16 max (v_pk_max_u16)
__device__ __forceinline__ float rowmax_bf16(const __bf16* rowp) {
  const uint4* row = (const uint4*)rowp;
  us2 m = (us2)0;
  #pragma unroll 4
  for (int q = 0; q < 16; ++q) {
    uint4 u = row[q];
    m = __builtin_elementwise_max(m, __builtin_bit_cast(us2, u.x));
    m = __builtin_elementwise_max(m, __builtin_bit_cast(us2, u.y));
    m = __builtin_elementwise_max(m, __builtin_bit_cast(us2, u.z));
    m = __builtin_elementwise_max(m, __builtin_bit_cast(us2, u.w));
  }
  const unsigned short h = m[0] > m[1] ? m[0] : m[1];
  return bf2f(h);
}

// ---------------------------------------------------------------------------
// Phase 1: each workgroup builds the log-semiring transfer matrix of one chunk
// of CHUNK time steps as (Q, s): G[i,j] = s[i] + log(Q[i,j]), Q row-normalized.
// LDS state U is kept in bf16; A-fragments are raw packed ds_load_b128's.
// Per step: U_{n+1} = diag(1/rowmax(U_n)) (U_n @ exp(trans)) diag(exp(e_t)).
// ---------------------------------------------------------------------------
__global__ __launch_bounds__(256) void crf_chunk_kernel(
    const float* __restrict__ emit, const float* __restrict__ trans,
    unsigned short* __restrict__ Gp, float* __restrict__ Gs)
{
  __shared__ __align__(16) __bf16 Ubf[NTS * SPB]; // unnormalized exp-domain state
  __shared__ float rinvS[NTS];                    // 1 / rowmax(U)
  __shared__ float svecS[NTS];                    // accumulated log row scales

  const int tid  = threadIdx.x;
  const int wave = tid >> 5;
  const int lane = tid & 31;
  const int hf   = lane >> 4;       // 0: lanes 0-15, 1: lanes 16-31
  const int l16  = lane & 15;
  const int col  = wave * 16 + l16; // this lane's output column N (0..127)

  const int c    = blockIdx.x;
  const int t0   = 1 + c * CHUNK;
  int tend = t0 + CHUNK; if (tend > TLEN) tend = TLEN;

  // B fragments: E = exp(trans), constant for the whole chunk (32 VGPRs).
  v16bf Bfrag[4];
  #pragma unroll
  for (int kc = 0; kc < 4; ++kc) {
    const int kbase = hf * 16 + kc * 32;
    #pragma unroll
    for (int e = 0; e < 16; ++e)
      Bfrag[kc][e] = (__bf16)expf(trans[(kbase + e) * NTS + col]);
  }
  if (tid < NTS) svecS[tid] = 0.0f;

  // Init with the first matrix of the chunk: U = exp(trans + emit[t0]).
  {
    const float* erow = emit + (size_t)t0 * NTS;
    for (int idx = tid; idx < NTS * NTS; idx += 256) {
      const int i = idx >> 7, j = idx & 127;
      Ubf[i * SPB + j] = (__bf16)expf(trans[idx] + erow[j]);
    }
  }
  __syncthreads();
  if (tid < NTS) {
    const float m = rowmax_bf16(Ubf + tid * SPB);
    rinvS[tid] = 1.0f / m;
    svecS[tid] += logf(m);
  }
  __syncthreads();

  // Main loop: one 128x128x128 matmul (bf16 WMMA, f32 accum) per time step.
  for (int t = t0 + 1; t < tend; ++t) {
    const float ee = expf(emit[(size_t)t * NTS + col]); // column scale exp(e_t[j])
    if (t + 8 < TLEN)
      __builtin_prefetch(emit + (size_t)(t + 8) * NTS + col, 0, 0);

    v8f acc[8];
    #pragma unroll
    for (int it = 0; it < 8; ++it) acc[it] = (v8f){0,0,0,0,0,0,0,0};

    #pragma unroll
    for (int it = 0; it < 8; ++it) {
      const __bf16* arow = Ubf + (it * 16 + l16) * SPB + hf * 16;
      #pragma unroll
      for (int kc = 0; kc < 4; ++kc) {
        const v8bf lo = *(const v8bf*)(arow + kc * 32);       // K = hf*16+32kc+0..7
        const v8bf hi = *(const v8bf*)(arow + kc * 32 + 8);   // K = +8..15
        const v16bf A = __builtin_shufflevector(lo, hi,
            0, 1, 2, 3, 4, 5, 6, 7, 8, 9, 10, 11, 12, 13, 14, 15);
        acc[it] = __builtin_amdgcn_wmma_f32_16x16x32_bf16(
            false, A, false, Bfrag[kc], (short)0, acc[it], false, false);
      }
    }
    __syncthreads();   // everyone done reading old U / old rinv is still live
    #pragma unroll
    for (int it = 0; it < 8; ++it) {
      const int r0 = it * 16 + hf * 8;
      const float4 ra = *(const float4*)(rinvS + r0);
      const float4 rb = *(const float4*)(rinvS + r0 + 4);
      __bf16* dst = Ubf + r0 * SPB + col;
      dst[0 * SPB] = (__bf16)(acc[it][0] * (ra.x * ee));
      dst[1 * SPB] = (__bf16)(acc[it][1] * (ra.y * ee));
      dst[2 * SPB] = (__bf16)(acc[it][2] * (ra.z * ee));
      dst[3 * SPB] = (__bf16)(acc[it][3] * (ra.w * ee));
      dst[4 * SPB] = (__bf16)(acc[it][4] * (rb.x * ee));
      dst[5 * SPB] = (__bf16)(acc[it][5] * (rb.y * ee));
      dst[6 * SPB] = (__bf16)(acc[it][6] * (rb.z * ee));
      dst[7 * SPB] = (__bf16)(acc[it][7] * (rb.w * ee));
    }
    __syncthreads();
    if (tid < NTS) {
      const float m = rowmax_bf16(Ubf + tid * SPB);
      rinvS[tid] = 1.0f / m;
      svecS[tid] += logf(m);
    }
    __syncthreads();
  }

  // Emit chunk result: normalized Q as bf16 bits + log scales.
  unsigned short* gp = Gp + (size_t)c * NTS * NTS;
  for (int idx = tid; idx < NTS * NTS; idx += 256) {
    const int i = idx >> 7, j = idx & 127;
    gp[idx] = bfbits((float)Ubf[i * SPB + j] * rinvS[i]);
  }
  if (tid < NTS) Gs[(size_t)c * NTS + tid] = svecS[tid];
}

// ---------------------------------------------------------------------------
// Tree combine: (Qa,sa) (x) (Qb,sb) -> slot a, in place (stride-2 slots).
// S = (Qa * diag(exp(sb-mb))) @ Qb  via bf16 WMMA, then row-renormalize.
// ---------------------------------------------------------------------------
__global__ __launch_bounds__(256) void crf_combine_kernel(
    unsigned short* __restrict__ Gp, float* __restrict__ Gs, int sstride)
{
  __shared__ float Pf[NTS * SPF];
  __shared__ float D[NTS];
  __shared__ float sash[NTS];
  __shared__ float rinv[NTS];
  __shared__ float mbsh;

  const int tid  = threadIdx.x;
  const int wave = tid >> 5;
  const int lane = tid & 31;
  const int hf   = lane >> 4;
  const int l16  = lane & 15;
  const int col  = wave * 16 + l16;

  const size_t a = (size_t)(2 * blockIdx.x) * (size_t)sstride;
  const size_t b = a + (size_t)sstride;
  const unsigned short* Pa = Gp + a * NTS * NTS;
  const unsigned short* Pb = Gp + b * NTS * NTS;

  if (tid < NTS) { sash[tid] = Gs[a * NTS + tid]; D[tid] = Gs[b * NTS + tid]; }
  __syncthreads();
  if (tid == 0) {
    float mm = -3.0e38f;
    for (int k = 0; k < NTS; ++k) mm = fmaxf(mm, D[k]);
    mbsh = mm;
  }
  __syncthreads();
  if (tid < NTS) D[tid] = expf(D[tid] - mbsh);
  __syncthreads();

  v8f acc[8];
  #pragma unroll
  for (int it = 0; it < 8; ++it) acc[it] = (v8f){0,0,0,0,0,0,0,0};

  #pragma unroll
  for (int kc = 0; kc < 4; ++kc) {
    v16bf B;
    const int kbase = hf * 16 + kc * 32;
    #pragma unroll
    for (int e = 0; e < 16; ++e)
      B[e] = __builtin_bit_cast(__bf16, Pb[(kbase + e) * NTS + col]);
    #pragma unroll 2
    for (int it = 0; it < 8; ++it) {
      const int arow = (it * 16 + l16) * NTS;
      v16bf A;
      const int kb0 = hf * 16 + kc * 32;
      #pragma unroll
      for (int e = 0; e < 8; ++e) {
        const int K = kb0 + e;
        A[e] = (__bf16)(bf2f(Pa[arow + K]) * D[K]);
      }
      #pragma unroll
      for (int e = 0; e < 8; ++e) {
        const int K = kb0 + 8 + e;
        A[e + 8] = (__bf16)(bf2f(Pa[arow + K]) * D[K]);
      }
      acc[it] = __builtin_amdgcn_wmma_f32_16x16x32_bf16(
          false, A, false, B, (short)0, acc[it], false, false);
    }
  }
  #pragma unroll
  for (int it = 0; it < 8; ++it)
    #pragma unroll
    for (int v = 0; v < 8; ++v)
      Pf[(it * 16 + hf * 8 + v) * SPF + col] = acc[it][v];
  __syncthreads();
  if (tid < NTS) {
    float m = 0.0f;
    const float4* row = (const float4*)(Pf + tid * SPF);
    for (int q = 0; q < NTS / 4; ++q) {
      float4 v = row[q];
      m = fmaxf(fmaxf(m, fmaxf(v.x, v.y)), fmaxf(v.z, v.w));
    }
    rinv[tid] = 1.0f / m;
    sash[tid] = sash[tid] + mbsh + logf(m);
  }
  __syncthreads();
  unsigned short* gp = Gp + a * NTS * NTS;
  for (int idx = tid; idx < NTS * NTS; idx += 256) {
    const int i = idx >> 7, j = idx & 127;
    gp[idx] = bfbits(Pf[i * SPF + j] * rinv[i]);
  }
  if (tid < NTS) Gs[a * NTS + tid] = sash[tid];
}

// ---------------------------------------------------------------------------
// Score: deterministic two-level reduction of trans[y_t,y_{t+1}] + emit[t,y_t]
// ---------------------------------------------------------------------------
__global__ __launch_bounds__(256) void crf_score_kernel(
    const float* __restrict__ emit, const int* __restrict__ y,
    const float* __restrict__ trans, float* __restrict__ partials)
{
  __shared__ float red[256];
  float acc = 0.0f;
  const int stride = gridDim.x * blockDim.x;
  for (int t = blockIdx.x * blockDim.x + threadIdx.x; t < TLEN - 1; t += stride) {
    const int yt = y[t], yn = y[t + 1];
    acc += trans[yt * NTS + yn] + emit[(size_t)t * NTS + yt];
  }
  red[threadIdx.x] = acc;
  __syncthreads();
  for (int o = 128; o > 0; o >>= 1) {
    if (threadIdx.x < o) red[threadIdx.x] += red[threadIdx.x + o];
    __syncthreads();
  }
  if (threadIdx.x == 0) partials[blockIdx.x] = red[0];
}

// ---------------------------------------------------------------------------
// Finish: logZ from (strans+emit[0]) (x) G (x) etrans, minus the path score.
// ---------------------------------------------------------------------------
__global__ __launch_bounds__(256) void crf_final_kernel(
    const unsigned short* __restrict__ P0, const float* __restrict__ S0,
    const float* __restrict__ emit, const int* __restrict__ y,
    const float* __restrict__ strans, const float* __restrict__ etrans,
    const float* __restrict__ partials, float* __restrict__ out)
{
  __shared__ float vrow[NTS];
  __shared__ float alphaJ[NTS];
  const int tid = threadIdx.x;
  if (tid < NTS) vrow[tid] = strans[tid] + emit[tid] + S0[tid]; // v_i + s_i
  __syncthreads();
  if (tid < NTS) {
    const int j = tid;
    float mm = -3.0e38f;
    for (int i = 0; i < NTS; ++i)
      mm = fmaxf(mm, vrow[i] + logf(bf2f(P0[i * NTS + j])));
    float ss = 0.0f;
    for (int i = 0; i < NTS; ++i)
      ss += expf(vrow[i] + logf(bf2f(P0[i * NTS + j])) - mm);
    alphaJ[j] = mm + logf(ss) + etrans[j];
  }
  __syncthreads();
  if (tid == 0) {
    float mm = -3.0e38f;
    for (int j = 0; j < NTS; ++j) mm = fmaxf(mm, alphaJ[j]);
    float ss = 0.0f;
    for (int j = 0; j < NTS; ++j) ss += expf(alphaJ[j] - mm);
    const float logZ = mm + logf(ss);
    float score = 0.0f;
    for (int p = 0; p < NSCORE; ++p) score += partials[p];
    const int y0 = y[0], yl = y[TLEN - 1];
    score += strans[y0] + etrans[yl] + emit[(size_t)(TLEN - 1) * NTS + yl];
    out[0] = logZ - score;
  }
}

extern "C" void kernel_launch(void* const* d_in, const int* in_sizes, int n_in,
                              void* d_out, int out_size, void* d_ws, size_t ws_size,
                              hipStream_t stream) {
  (void)in_sizes; (void)n_in; (void)out_size; (void)ws_size;
  const float* emit   = (const float*)d_in[0];
  const int*   y      = (const int*)d_in[1];
  const float* trans  = (const float*)d_in[2];
  const float* strans = (const float*)d_in[3];
  const float* etrans = (const float*)d_in[4];
  float* out = (float*)d_out;

  char* ws = (char*)d_ws;
  unsigned short* Gp = (unsigned short*)ws;                    // 1024*128*128*2 = 32 MiB
  float* Gs       = (float*)(ws + (size_t)NCHUNK * NTS * NTS * 2);     // 512 KiB
  float* partials = (float*)(ws + (size_t)NCHUNK * NTS * NTS * 2
                                + (size_t)NCHUNK * NTS * 4);           // 2 KiB

  crf_chunk_kernel<<<NCHUNK, 256, 0, stream>>>(emit, trans, Gp, Gs);
  crf_score_kernel<<<NSCORE, 256, 0, stream>>>(emit, y, trans, partials);

  int s = 1;
  for (int n = NCHUNK / 2; n >= 1; n >>= 1) {
    crf_combine_kernel<<<n, 256, 0, stream>>>(Gp, Gs, s);
    s <<= 1;
  }
  crf_final_kernel<<<1, 256, 0, stream>>>(Gp, Gs, emit, y, strans, etrans, partials, out);
}
